// LSTM_74543452389909
// MI455X (gfx1250) — matrix operand order repaired and checked by reading, not compiled, not run on hardware
//
#include <hip/hip_runtime.h>

// ---------------------------------------------------------------------------
// LSTM, N=64, D=2048, IN=256, H=512  — persistent bf16-WMMA kernel for gfx1250
// Pipeline per step: x-part WMMA (no h dep) -> per-wave barrier spin ->
// async h -> h-part WMMA -> pointwise -> arrive.
// ---------------------------------------------------------------------------

typedef __attribute__((ext_vector_type(16))) __bf16 v16bf;
typedef __attribute__((ext_vector_type(8)))  float  v8f;

#define N_    64
#define D_    2048
#define IN_   256
#define H_    512
#define K_    768            // IN_ + H_
#define GC_   2048           // 4 * H_ gate columns (order: i, f, g, o)
#define NWG_  32             // persistent workgroups
#define JW_   16             // hidden units per WG
#define CW_   64             // local gate columns per WG (4 gates x 16)
#define XS_   264            // LDS row stride (elems) for x tile  (256 + 8)
#define HS_   520            // LDS row stride (elems) for h tile  (512 + 8)
#define CS_   776            // LDS col stride (elems) for weights (768 + 8)

// LDS layout (bytes)
#define XL0_OFF   0u
#define XL1_OFF   33792u
#define HL_OFF    67584u
#define WL_OFF    134144u
#define GL_OFF    233472u
#define CL_OFF    249856u
#define BL_OFF    253952u
#define LDS_BYTES 254208u

// workspace layout (bytes)
#define WPACK_OFF 0u                               // bf16 weights, col-major [2048][768]
#define BIAS_OFF  (GC_ * K_ * 2u)                  // fused fp32 bias [2048]
#define HB0_OFF   (BIAS_OFF + GC_ * 4u)            // h buffer 0, bf16 [64][512]
#define HB1_OFF   (HB0_OFF + N_ * H_ * 2u)         // h buffer 1
#define CNT_OFF   (HB1_OFF + N_ * H_ * 2u)         // barrier counter (u32)

union Frag { v16bf v; uint4 u[2]; };

__device__ __forceinline__ unsigned short f2bf(float f) {   // round-to-nearest-even
  unsigned u = __float_as_uint(f);
  u += 0x7FFFu + ((u >> 16) & 1u);
  return (unsigned short)(u >> 16);
}
__device__ __forceinline__ float sigm(float x)  { return 1.0f / (1.0f + __expf(-x)); }
__device__ __forceinline__ float tanh_(float x) { return 1.0f - 2.0f / (__expf(2.0f * x) + 1.0f); }

// ---------------------------------------------------------------------------
// Prep: fuse/pack weights to bf16 column-major, fuse biases, zero h0, reset ctr
// ---------------------------------------------------------------------------
__global__ void lstm_prep(const float* Wii, const float* Wif, const float* Wig, const float* Wio,
                          const float* Whi, const float* Whf, const float* Whg, const float* Who,
                          const float* bii, const float* bhi, const float* bif, const float* bhf,
                          const float* big, const float* bhg, const float* bio, const float* bho,
                          unsigned char* ws)
{
  unsigned short* wpack = (unsigned short*)(ws + WPACK_OFF);
  float*          bias  = (float*)(ws + BIAS_OFF);
  unsigned short* hb0   = (unsigned short*)(ws + HB0_OFF);
  unsigned*       cnt   = (unsigned*)(ws + CNT_OFF);

  const float* Wi[4] = {Wii, Wif, Wig, Wio};
  const float* Wh[4] = {Whi, Whf, Whg, Who};
  const float* Bi[4] = {bii, bif, big, bio};
  const float* Bh[4] = {bhi, bhf, bhg, bho};

  int gid = blockIdx.x * blockDim.x + threadIdx.x;
  int gsz = gridDim.x * blockDim.x;

  for (int i = gid; i < GC_ * K_; i += gsz) {
    int col = i / K_, k = i - col * K_;
    int g = col >> 9, j = col & 511;
    float v = (k < IN_) ? Wi[g][k * H_ + j] : Wh[g][(k - IN_) * H_ + j];
    wpack[col * K_ + k] = f2bf(v);
  }
  for (int i = gid; i < GC_; i += gsz) {
    int g = i >> 9, j = i & 511;
    bias[i] = Bi[g][j] + Bh[g][j];
  }
  for (int i = gid; i < N_ * H_; i += gsz) hb0[i] = 0;
  if (gid == 0) *cnt = 0u;
}

// ---------------------------------------------------------------------------
// Main persistent kernel: 32 WGs x 256 threads (8 wave32), all 2048 steps
// ---------------------------------------------------------------------------
__global__ void __launch_bounds__(256, 1)
lstm_main(const float* __restrict__ x, unsigned char* __restrict__ ws, float* __restrict__ out)
{
  extern __shared__ unsigned char smem[];
  unsigned short* xl0   = (unsigned short*)(smem + XL0_OFF);   // x tile, buffer 0
  unsigned short* xl1   = (unsigned short*)(smem + XL1_OFF);   // x tile, buffer 1
  unsigned short* hl    = (unsigned short*)(smem + HL_OFF);    // h tile
  unsigned short* wlds  = (unsigned short*)(smem + WL_OFF);    // weights (col-major, padded)
  float*          glds  = (float*)(smem + GL_OFF);             // gates [64 cols][64 rows]
  float*          clds  = (float*)(smem + CL_OFF);             // c state [16 j][64 n]
  float*          biasl = (float*)(smem + BL_OFF);

  const unsigned short* wpack = (const unsigned short*)(ws + WPACK_OFF);
  const float*          bias  = (const float*)(ws + BIAS_OFF);
  unsigned short*       hb0   = (unsigned short*)(ws + HB0_OFF);
  unsigned short*       hb1   = (unsigned short*)(ws + HB1_OFF);
  unsigned*             cnt   = (unsigned*)(ws + CNT_OFF);

  const int tid  = threadIdx.x;
  const int wg   = blockIdx.x;
  const int wave = tid >> 5;
  const int lane = tid & 31;
  const int lm   = lane & 15;
  const int half = lane >> 4;

  // ---- one-time: stage this WG's weight slice into LDS (col-major, padded)
  for (int i = tid; i < CW_ * K_; i += 256) {
    int cl = i / K_, k = i - cl * K_;
    int gate = cl >> 4, jl = cl & 15;
    int gcol = gate * H_ + wg * JW_ + jl;
    wlds[cl * CS_ + k] = wpack[gcol * K_ + k];
  }
  for (int i = tid; i < CW_; i += 256) {
    int gate = i >> 4, jl = i & 15;
    biasl[i] = bias[gate * H_ + wg * JW_ + jl];
  }
  for (int i = tid; i < JW_ * N_; i += 256) clds[i] = 0.0f;

  // ---- stage x_0 (fp32 -> bf16) into buffer 0
  {
    const float* xt = x;
    for (int i = tid; i < (N_ * IN_) / 4; i += 256) {
      int n = i >> 6, k4 = (i & 63) * 4;
      float4 f = *(const float4*)(xt + (size_t)n * (D_ * IN_) + k4);
      unsigned s0 = f2bf(f.x), s1 = f2bf(f.y), s2 = f2bf(f.z), s3 = f2bf(f.w);
      *(uint2*)(xl0 + n * XS_ + k4) = make_uint2(s0 | (s1 << 16), s2 | (s3 << 16));
    }
  }
  __syncthreads();   // wlds / xl0 / clds visible to all waves

  // LDS byte address of hl (low 32 bits of flat shared pointer = LDS offset)
  const unsigned hl_base = (unsigned)(unsigned long long)(void*)hl;

  // wave -> two 16x16 output tiles sharing one row-tile (A fragment reuse)
  const int rt  = wave >> 1;          // row tile 0..3 (batch)
  const int ct0 = (wave & 1) * 2;     // local col tiles ct0, ct0+1
  const int row = rt * 16 + lm;       // this lane's A-matrix row (batch index)
  const int cl0 = ct0 * 16 + lm;      // B/D column (local) for tile 0
  const int cl1 = cl0 + 16;           // for tile 1

  for (int t = 0; t < D_; ++t) {
    const unsigned short* hsrc = (t & 1) ? hb1 : hb0;
    unsigned short*       hbn  = (t & 1) ? hb0 : hb1;
    const unsigned short* xl   = (t & 1) ? xl1 : xl0;

    // ================= x-part GEMM: no dependence on h; runs while other
    // WGs are still finishing step t-1 (hides barrier skew) =================
    v8f a0a = {}, a0b = {}, a1a = {}, a1b = {};

    #pragma unroll
    for (int kb = 0; kb < 8; ++kb) {            // K = 0..255
      Frag a, b0, b1;
      const unsigned short* pa  = xl + row * XS_ + kb * 32 + half * 8;
      a.u[0]  = *(const uint4*)pa;  a.u[1]  = *(const uint4*)(pa + 16);
      const unsigned short* pb0 = wlds + cl0 * CS_ + kb * 32 + half * 8;
      b0.u[0] = *(const uint4*)pb0; b0.u[1] = *(const uint4*)(pb0 + 16);
      const unsigned short* pb1 = wlds + cl1 * CS_ + kb * 32 + half * 8;
      b1.u[0] = *(const uint4*)pb1; b1.u[1] = *(const uint4*)(pb1 + 16);
      if (kb & 1) {
        a0b = __builtin_amdgcn_wmma_f32_16x16x32_bf16(false, a.v, false, b0.v, (short)0, a0b, false, false);
        a1b = __builtin_amdgcn_wmma_f32_16x16x32_bf16(false, a.v, false, b1.v, (short)0, a1b, false, false);
      } else {
        a0a = __builtin_amdgcn_wmma_f32_16x16x32_bf16(false, a.v, false, b0.v, (short)0, a0a, false, false);
        a1a = __builtin_amdgcn_wmma_f32_16x16x32_bf16(false, a.v, false, b1.v, (short)0, a1a, false, false);
      }
    }

    // ================= per-wave wait for h_{t} release (whole wave stalls
    // on the divergent lane-0 spin), then async-load h straight into LDS ====
    if (lane == 0) {
      unsigned target = (unsigned)t * NWG_;
      while (__hip_atomic_load(cnt, __ATOMIC_ACQUIRE, __HIP_MEMORY_SCOPE_AGENT) < target) {
        __builtin_amdgcn_s_sleep(1);
      }
    }
    __builtin_amdgcn_fence(__ATOMIC_ACQUIRE, "agent");
    {
      unsigned long long src = (unsigned long long)hsrc;
      for (int i = tid; i < (N_ * H_) / 8; i += 256) {
        int n = i >> 6, q = i & 63;
        unsigned voff = (unsigned)i * 16u;                       // byte offset in hsrc
        unsigned ldsa = hl_base + (unsigned)(n * HS_ * 2 + q * 16);
        asm volatile("global_load_async_to_lds_b128 %0, %1, %2"
                     :: "v"(ldsa), "v"(voff), "s"(src) : "memory");
      }
      asm volatile("s_wait_asynccnt 0" ::: "memory");
    }
    __syncthreads();   // hl complete across all waves

    // ================= h-part GEMM: K = 256..767 ===========================
    #pragma unroll
    for (int kb = 0; kb < 16; ++kb) {
      Frag a, b0, b1;
      const unsigned short* pa  = hl + row * HS_ + kb * 32 + half * 8;
      a.u[0]  = *(const uint4*)pa;  a.u[1]  = *(const uint4*)(pa + 16);
      const unsigned short* pb0 = wlds + cl0 * CS_ + 256 + kb * 32 + half * 8;
      b0.u[0] = *(const uint4*)pb0; b0.u[1] = *(const uint4*)(pb0 + 16);
      const unsigned short* pb1 = wlds + cl1 * CS_ + 256 + kb * 32 + half * 8;
      b1.u[0] = *(const uint4*)pb1; b1.u[1] = *(const uint4*)(pb1 + 16);
      if (kb & 1) {
        a0b = __builtin_amdgcn_wmma_f32_16x16x32_bf16(false, a.v, false, b0.v, (short)0, a0b, false, false);
        a1b = __builtin_amdgcn_wmma_f32_16x16x32_bf16(false, a.v, false, b1.v, (short)0, a1b, false, false);
      } else {
        a0a = __builtin_amdgcn_wmma_f32_16x16x32_bf16(false, a.v, false, b0.v, (short)0, a0a, false, false);
        a1a = __builtin_amdgcn_wmma_f32_16x16x32_bf16(false, a.v, false, b1.v, (short)0, a1a, false, false);
      }
    }
    v8f acc0 = a0a + a0b;
    v8f acc1 = a1a + a1b;

    // ---- bias + D-tile -> glds  (D layout: elem r => row rt*16+half*8+r, col lm)
    {
      float b0v = biasl[cl0], b1v = biasl[cl1];
      int rbase = rt * 16 + half * 8;
      *(float4*)&glds[cl0 * 64 + rbase]     = make_float4(acc0[0]+b0v, acc0[1]+b0v, acc0[2]+b0v, acc0[3]+b0v);
      *(float4*)&glds[cl0 * 64 + rbase + 4] = make_float4(acc0[4]+b0v, acc0[5]+b0v, acc0[6]+b0v, acc0[7]+b0v);
      *(float4*)&glds[cl1 * 64 + rbase]     = make_float4(acc1[0]+b1v, acc1[1]+b1v, acc1[2]+b1v, acc1[3]+b1v);
      *(float4*)&glds[cl1 * 64 + rbase + 4] = make_float4(acc1[4]+b1v, acc1[5]+b1v, acc1[6]+b1v, acc1[7]+b1v);
    }
    __syncthreads();

    // ---- prefetch x_{t+1} into the other buffer; overlaps pointwise
    if (t + 1 < D_) {
      unsigned short* xn = (t & 1) ? xl0 : xl1;
      const float* xt = x + (size_t)(t + 1) * IN_;
      for (int i = tid; i < (N_ * IN_) / 4; i += 256) {
        int n = i >> 6, k4 = (i & 63) * 4;
        float4 f = *(const float4*)(xt + (size_t)n * (D_ * IN_) + k4);
        unsigned s0 = f2bf(f.x), s1 = f2bf(f.y), s2 = f2bf(f.z), s3 = f2bf(f.w);
        *(uint2*)(xn + n * XS_ + k4) = make_uint2(s0 | (s1 << 16), s2 | (s3 << 16));
      }
    }

    // ---- pointwise LSTM cell update (1024 (n,j) values, 4 per thread)
    bool last = (t == D_ - 1);
    #pragma unroll
    for (int rep = 0; rep < 4; ++rep) {
      int idx = rep * 256 + tid;
      int n = idx & 63, jl = idx >> 6;
      float iv = glds[( 0 + jl) * 64 + n];
      float fv = glds[(16 + jl) * 64 + n];
      float gv = glds[(32 + jl) * 64 + n];
      float ov = glds[(48 + jl) * 64 + n];
      float c  = sigm(fv) * clds[idx] + sigm(iv) * tanh_(gv);
      float h  = sigm(ov) * tanh_(c);
      clds[idx] = c;
      int J = wg * JW_ + jl;
      hbn[n * H_ + J] = f2bf(h);
      if (last) {
        out[n * H_ + J]           = h;   // h_t
        out[N_ * H_ + n * H_ + J] = c;   // c_t
      }
    }

    // ---- release h_{t+1}: fence own stores, WG-sync, single arrive (no spin)
    __builtin_amdgcn_fence(__ATOMIC_RELEASE, "agent");
    __syncthreads();
    if (tid == 0) atomicAdd(cnt, 1u);
  }
}

// ---------------------------------------------------------------------------
extern "C" void kernel_launch(void* const* d_in, const int* in_sizes, int n_in,
                              void* d_out, int out_size, void* d_ws, size_t ws_size,
                              hipStream_t stream)
{
  (void)in_sizes; (void)n_in; (void)out_size; (void)ws_size;
  const float* x = (const float*)d_in[0];

  lstm_prep<<<1024, 256, 0, stream>>>(
      (const float*)d_in[1],  (const float*)d_in[2],  (const float*)d_in[3],  (const float*)d_in[4],
      (const float*)d_in[5],  (const float*)d_in[6],  (const float*)d_in[7],  (const float*)d_in[8],
      (const float*)d_in[9],  (const float*)d_in[10], (const float*)d_in[11], (const float*)d_in[12],
      (const float*)d_in[13], (const float*)d_in[14], (const float*)d_in[15], (const float*)d_in[16],
      (unsigned char*)d_ws);

  lstm_main<<<NWG_, 256, LDS_BYTES, stream>>>(x, (unsigned char*)d_ws, (float*)d_out);
}
